// PerformerTransformer_12945031430443
// MI455X (gfx1250) — compile-verified, matrix operand order checked by reading, not run on hardware
//
#include <hip/hip_runtime.h>
#include <hip/hip_bf16.h>

// ---------------- types ----------------
typedef __attribute__((ext_vector_type(16))) __bf16          bf16x16;
typedef __attribute__((ext_vector_type(8)))  unsigned short  ushortx8;
typedef __attribute__((ext_vector_type(8)))  float           floatx8;

// ---------------- constants ----------------
constexpr int B_    = 2;
constexpr int N_    = 16384;
constexpr int D_    = 256;
constexpr int H_    = 4;
constexpr int DH_   = 64;
constexpr int MF_   = 266;   // true feature count
constexpr int MP_   = 288;   // padded to multiple of 32 (no K-tail anywhere)
constexpr int MLP_  = 1024;
constexpr int DEPTH_= 2;
constexpr int ROWS_ = B_ * N_;          // 32768
constexpr int LDM_  = H_ * MP_;         // 1152
constexpr float LN_EPS   = 1e-5f;
constexpr float KERN_EPS = 1e-3f;
constexpr float NORMER   = 0.3535533906f;   // 64^-0.25
constexpr float RATIO    = 0.06131393f;     // 266^-0.5

// ---------------- bf16 helpers ----------------
__device__ __forceinline__ float bf2f(unsigned short u) {
  union { unsigned int i; float f; } x; x.i = ((unsigned int)u) << 16; return x.f;
}
__device__ __forceinline__ unsigned short f2bf(float f) {
  union { unsigned int i; float f; } x; x.f = f;
  unsigned int r = x.i + 0x7FFFu + ((x.i >> 16) & 1u);
  return (unsigned short)(r >> 16);
}
__device__ __forceinline__ unsigned lds_off(const void* p) {
  // generic LDS pointer low 32 bits == byte offset within workgroup LDS
  return (unsigned)(unsigned long long)(uintptr_t)p;
}
// async DMA: global -> LDS, 16 bytes per lane, tracked by ASYNCcnt
__device__ __forceinline__ void async_b128(unsigned dst_lds, const void* src) {
  asm volatile("global_load_async_to_lds_b128 %0, %1, off"
               :: "v"(dst_lds), "v"(src) : "memory");
}
__device__ __forceinline__ void wait_async0() {
  asm volatile("s_wait_asynccnt 0" ::: "memory");
}

// ---------------- fragment loader (16-bit A/B WMMA layout) ----------------
// rowp = &tile[row0 + lane%16][0], stride-40 padded rows (16B aligned).
// lanes 0-15: K = 0..7 / 16..23 ; lanes 16-31: K = 8..15 / 24..31
__device__ __forceinline__ bf16x16 load_frag(const unsigned short* rowp, int lane) {
  int kb = (lane >> 4) << 3;
  ushortx8 lo = *(const ushortx8*)(rowp + kb);        // ds_load_b128
  ushortx8 hi = *(const ushortx8*)(rowp + kb + 16);   // ds_load_b128
  return __builtin_bit_cast(bf16x16,
      __builtin_shufflevector(lo, hi, 0, 1, 2, 3, 4, 5, 6, 7,
                              8, 9, 10, 11, 12, 13, 14, 15));
}

// ---------------- generic WMMA bf16 GEMM ----------------
// A row-major [M][K] (lda), Bt K-MAJOR (transposed) [N][K] (ldbt).
// REQUIRES: K % 32 == 0; all row strides multiple of 8 elements; bases 16B aligned.
// 128x64 block tile, 8 waves, 2x2 WMMA tiles/wave, K-step 32,
// double-buffered LDS fed by async global->LDS DMA (ASYNCcnt).
// C = act(scale*A@Bt' + bias) + resid ; outputs: f32 [M][N], bf16 [M][N], bf16T [N][M]
__global__ __launch_bounds__(256) void k_gemm(
    const unsigned short* __restrict__ A, int lda,
    const unsigned short* __restrict__ Bt, int ldbt,
    const float* __restrict__ bias,
    const float* __restrict__ resid,
    float* __restrict__ outF,
    unsigned short* __restrict__ outB,
    unsigned short* __restrict__ outBT, int ldt,
    int ldc, int M, int N, int K, float scale, int act)
{
  __shared__ unsigned short As[2][128][40];   // [m][k], 80B padded rows
  __shared__ unsigned short Bs[2][64][40];    // [n][k], 80B padded rows

  const int tid  = threadIdx.x;
  const int lane = tid & 31;
  const int w    = tid >> 5;        // 8 waves
  const int wr   = w & 3;           // M sub-block rows wr*32..+31
  const int wc   = w >> 2;          // N sub-block cols wc*32..+31
  const int rowBase = blockIdx.y * 128;
  const int colBase = blockIdx.x * 64;

  // staging maps (branch-free; rows clamped in-bounds, garbage masked at store)
  const int ar0  = tid >> 2;                 // A: rows ar0, ar0+64; seg aseg
  const int aseg = tid & 3;
  const int bn   = tid >> 2;                 // B: row bn (n), seg bseg
  const int bseg = tid & 3;

  const int gra = min(rowBase + ar0, M - 1);
  const int grb = min(rowBase + ar0 + 64, M - 1);
  const int gbn = min(colBase + bn, N - 1);

  floatx8 acc[2][2] = {};

  const unsigned a0dst = lds_off(&As[0][ar0][aseg * 8]);
  const unsigned a1dst = lds_off(&As[0][ar0 + 64][aseg * 8]);
  const unsigned bdst  = lds_off(&Bs[0][bn][bseg * 8]);
  const unsigned abuf  = sizeof(unsigned short) * 128 * 40;   // As buffer stride
  const unsigned bbuf  = sizeof(unsigned short) * 64 * 40;    // Bs buffer stride

  auto stage_async = [&](int buf, int kk) {
    const unsigned short* pa0 = A  + (size_t)gra * lda  + kk + aseg * 8;
    const unsigned short* pa1 = A  + (size_t)grb * lda  + kk + aseg * 8;
    const unsigned short* pb  = Bt + (size_t)gbn * ldbt + kk + bseg * 8;
    async_b128(a0dst + buf * abuf, pa0);
    async_b128(a1dst + buf * abuf, pa1);
    async_b128(bdst  + buf * bbuf, pb);
  };

  const int nk = K >> 5;
  stage_async(0, 0);
  wait_async0();
  __syncthreads();

  for (int t = 0; t < nk; ++t) {
    const int cur = t & 1;
    const bool more = (t + 1) < nk;
    if (more) stage_async(cur ^ 1, (t + 1) * 32);   // DMA overlaps WMMA below

    bf16x16 a0 = load_frag(&As[cur][wr * 32 +      (lane & 15)][0], lane);
    bf16x16 a1 = load_frag(&As[cur][wr * 32 + 16 + (lane & 15)][0], lane);
    bf16x16 b0 = load_frag(&Bs[cur][wc * 32 +      (lane & 15)][0], lane);
    bf16x16 b1 = load_frag(&Bs[cur][wc * 32 + 16 + (lane & 15)][0], lane);

    acc[0][0] = __builtin_amdgcn_wmma_f32_16x16x32_bf16(false, a0, false, b0,
                                                        (short)0, acc[0][0], false, false);
    acc[0][1] = __builtin_amdgcn_wmma_f32_16x16x32_bf16(false, a0, false, b1,
                                                        (short)0, acc[0][1], false, false);
    acc[1][0] = __builtin_amdgcn_wmma_f32_16x16x32_bf16(false, a1, false, b0,
                                                        (short)0, acc[1][0], false, false);
    acc[1][1] = __builtin_amdgcn_wmma_f32_16x16x32_bf16(false, a1, false, b1,
                                                        (short)0, acc[1][1], false, false);

    if (more) wait_async0();
    __syncthreads();
  }

  const int half = lane >> 4, lr = lane & 15;
#pragma unroll
  for (int i = 0; i < 2; ++i) {
#pragma unroll
    for (int j = 0; j < 2; ++j) {
      int gn = colBase + wc * 32 + j * 16 + lr;
#pragma unroll
      for (int r = 0; r < 8; ++r) {
        int gm = rowBase + wr * 32 + i * 16 + half * 8 + r;
        if (gm < M && gn < N) {
          float v = acc[i][j][r] * scale;
          if (bias) v += bias[gn];
          if (act == 1) v = 0.5f * v * (1.0f + erff(v * 0.70710678f));  // exact gelu
          if (resid) v += resid[(size_t)gm * ldc + gn];
          if (outF)  outF[(size_t)gm * ldc + gn] = v;
          if (outB)  outB[(size_t)gm * ldc + gn] = f2bf(v);
          if (outBT) outBT[(size_t)gn * ldt + gm] = f2bf(v);
        }
      }
    }
  }
}

// ---------------- layernorm: one row per block, bf16 out ----------------
__global__ __launch_bounds__(256) void k_ln(const float* __restrict__ x,
                                            const float* __restrict__ w,
                                            const float* __restrict__ b,
                                            unsigned short* __restrict__ out) {
  __shared__ float red[256];
  int row = blockIdx.x, t = threadIdx.x;
  float v = x[(size_t)row * 256 + t];
  red[t] = v; __syncthreads();
  for (int s = 128; s; s >>= 1) { if (t < s) red[t] += red[t + s]; __syncthreads(); }
  float mu = red[0] * (1.0f / 256.0f);
  __syncthreads();
  float d = v - mu;
  red[t] = d * d; __syncthreads();
  for (int s = 128; s; s >>= 1) { if (t < s) red[t] += red[t + s]; __syncthreads(); }
  float var = red[0] * (1.0f / 256.0f);
  out[(size_t)row * 256 + t] = f2bf(d * rsqrtf(var + LN_EPS) * w[t] + b[t]);
}

// ---------------- f32 [K][N] -> bf16 TRANSPOSED [N][K] ----------------
__global__ void k_cvtT(const float* __restrict__ src, unsigned short* __restrict__ dst,
                       int K, int N) {
  int i = blockIdx.x * 256 + threadIdx.x;
  if (i >= K * N) return;
  int n = i / K, k = i % K;
  dst[i] = f2bf(src[(size_t)k * N + n]);
}

// ---------------- projT: [266][64] f32 -> [288][64] bf16 m-major (zero padded) ----------------
__global__ void k_projT(const float* __restrict__ proj, unsigned short* __restrict__ out) {
  int i = blockIdx.x * 256 + threadIdx.x;
  if (i >= MP_ * DH_) return;
  int m = i >> 6, d = i & 63;
  out[i] = (m < MF_) ? f2bf(proj[m * DH_ + d]) : (unsigned short)0;
}

// ---------------- diag: 0.5 * sum(qn^2) per (row, head) ----------------
__global__ void k_diag(const unsigned short* __restrict__ q, float* __restrict__ diag) {
  int rh = blockIdx.x * 256 + threadIdx.x;
  if (rh >= ROWS_ * H_) return;
  int row = rh >> 2, h = rh & 3;
  const unsigned short* p = q + (size_t)row * D_ + h * DH_;
  float s = 0.f;
  for (int j = 0; j < DH_; ++j) { float v = bf2f(p[j]); s += v * v; }
  diag[rh] = 0.5f * s;
}

// ---------------- query exp-kernel: rowwise stab, write qp bf16 (padded zeros) ----------------
__global__ __launch_bounds__(128) void k_exp_q(const float* __restrict__ dash,
                                               const float* __restrict__ diag,
                                               unsigned short* __restrict__ qp) {
  __shared__ float red[128];
  int rh = blockIdx.x;                 // row*4 + h
  int row = rh >> 2, h = rh & 3, t = threadIdx.x;
  const float* drow = dash + (size_t)row * LDM_ + h * MP_;
  float m = -1e30f;
  for (int i = t; i < MF_; i += 128) m = fmaxf(m, drow[i]);
  red[t] = m; __syncthreads();
  for (int s = 64; s; s >>= 1) { if (t < s) red[t] = fmaxf(red[t], red[t + s]); __syncthreads(); }
  float stab = red[0];
  float dg = diag[rh];
  unsigned short* qrow = qp + (size_t)row * LDM_ + h * MP_;
  for (int i = t; i < MP_; i += 128) {
    float v = (i < MF_) ? RATIO * (expf(drow[i] - dg - stab) + KERN_EPS) : 0.f;
    qrow[i] = f2bf(v);
  }
}

// ---------------- key global max, two passes ----------------
__global__ __launch_bounds__(256) void k_maxp(const float* __restrict__ dash,
                                              float* __restrict__ part, size_t total) {
  __shared__ float red[256];
  float m = -1e30f;
  for (size_t i = (size_t)blockIdx.x * 256 + threadIdx.x; i < total;
       i += (size_t)gridDim.x * 256) {
    int col = (int)(i % LDM_);
    if ((col % MP_) < MF_) m = fmaxf(m, dash[i]);
  }
  red[threadIdx.x] = m; __syncthreads();
  for (int s = 128; s; s >>= 1) {
    if (threadIdx.x < s) red[threadIdx.x] = fmaxf(red[threadIdx.x], red[threadIdx.x + s]);
    __syncthreads();
  }
  if (threadIdx.x == 0) part[blockIdx.x] = red[0];
}
__global__ __launch_bounds__(256) void k_maxf(const float* __restrict__ part,
                                              float* __restrict__ gmax, int n) {
  __shared__ float red[256];
  float m = -1e30f;
  for (int i = threadIdx.x; i < n; i += 256) m = fmaxf(m, part[i]);
  red[threadIdx.x] = m; __syncthreads();
  for (int s = 128; s; s >>= 1) {
    if (threadIdx.x < s) red[threadIdx.x] = fmaxf(red[threadIdx.x], red[threadIdx.x + s]);
    __syncthreads();
  }
  if (threadIdx.x == 0) gmax[0] = red[0];
}

// ---------------- key exp-kernel: global stab, write kp TRANSPOSED [bh][m][n] ----------------
__global__ __launch_bounds__(128) void k_exp_k(const float* __restrict__ dash,
                                               const float* __restrict__ diag,
                                               const float* __restrict__ gmax,
                                               unsigned short* __restrict__ kpT) {
  int rh = blockIdx.x;
  int row = rh >> 2, h = rh & 3;
  int b = row >> 14, n = row & (N_ - 1);
  const float* drow = dash + (size_t)row * LDM_ + h * MP_;
  float stab = gmax[0], dg = diag[rh];
  size_t base = ((size_t)(b * H_ + h) * MP_) * N_ + n;
  for (int i = threadIdx.x; i < MP_; i += 128) {
    float v = (i < MF_) ? RATIO * (expf(drow[i] - dg - stab) + KERN_EPS) : 0.f;
    kpT[base + (size_t)i * N_] = f2bf(v);
  }
}

// ---------------- k_sum: sum kpT row over n (contiguous) ----------------
__global__ __launch_bounds__(256) void k_ksum(const unsigned short* __restrict__ kpT,
                                              float* __restrict__ ksum) {
  __shared__ float red[256];
  int bm = blockIdx.x;                       // bh*MP_ + m
  const unsigned short* p = kpT + (size_t)bm * N_;
  float s = 0.f;
  for (int i = threadIdx.x; i < N_; i += 256) s += bf2f(p[i]);
  red[threadIdx.x] = s; __syncthreads();
  for (int st = 128; st; st >>= 1) {
    if (threadIdx.x < st) red[threadIdx.x] += red[threadIdx.x + st];
    __syncthreads();
  }
  if (threadIdx.x == 0) ksum[bm] = red[0];
}

// ---------------- d_inv = 1 / (qp . ksum) ----------------
__global__ void k_dinv(const unsigned short* __restrict__ qp,
                       const float* __restrict__ ksum, float* __restrict__ dinv) {
  int rh = blockIdx.x * 256 + threadIdx.x;
  if (rh >= ROWS_ * H_) return;
  int row = rh >> 2, h = rh & 3;
  int b = row >> 14;
  const unsigned short* qrow = qp + (size_t)row * LDM_ + h * MP_;
  const float* ks = ksum + (b * H_ + h) * MP_;
  float s = 0.f;
  for (int i = 0; i < MP_; ++i) s += bf2f(qrow[i]) * ks[i];
  dinv[rh] = 1.0f / s;
}

// ---------------- scale attention output by d_inv, emit bf16 ----------------
__global__ void k_scale(const float* __restrict__ att, const float* __restrict__ dinv,
                        unsigned short* __restrict__ attb) {
  size_t i = (size_t)blockIdx.x * 256 + threadIdx.x;
  if (i >= (size_t)ROWS_ * D_) return;
  int row = (int)(i >> 8), col = (int)(i & 255);
  attb[i] = f2bf(att[i] * dinv[row * H_ + (col >> 6)]);
}

// ---------------- host side ----------------
static inline void gemm(hipStream_t s, const unsigned short* A, int lda,
                        const unsigned short* Bt, int ldbt,
                        const float* bias, const float* resid,
                        float* outF, unsigned short* outB,
                        unsigned short* outBT, int ldt,
                        int ldc, int M, int N, int K, float scale, int act) {
  dim3 g((N + 63) / 64, (M + 127) / 128);
  k_gemm<<<g, 256, 0, s>>>(A, lda, Bt, ldbt, bias, resid, outF, outB, outBT, ldt,
                           ldc, M, N, K, scale, act);
}

extern "C" void kernel_launch(void* const* d_in, const int* in_sizes, int n_in,
                              void* d_out, int out_size, void* d_ws, size_t ws_size,
                              hipStream_t stream) {
  (void)in_sizes; (void)n_in; (void)out_size; (void)ws_size;

  const float* x_in = (const float*)d_in[0];
  const float* ln1w = (const float*)d_in[1];
  const float* ln1b = (const float*)d_in[2];
  const float* wq   = (const float*)d_in[3];
  const float* wk   = (const float*)d_in[4];
  const float* wv   = (const float*)d_in[5];
  const float* wo   = (const float*)d_in[6];
  const float* bo   = (const float*)d_in[7];
  const float* ln2w = (const float*)d_in[8];
  const float* ln2b = (const float*)d_in[9];
  const float* w1   = (const float*)d_in[10];
  const float* b1   = (const float*)d_in[11];
  const float* w2   = (const float*)d_in[12];
  const float* b2   = (const float*)d_in[13];
  const float* proj = (const float*)d_in[14];

  // ---- workspace carve-out (256B aligned) ----
  char* ws = (char*)d_ws;
  size_t off = 0;
  auto take = [&](size_t bytes) -> char* {
    char* p = ws + off;
    off = (off + bytes + 255) & ~(size_t)255;
    return p;
  };
  float*          xbuf = (float*)         take((size_t)ROWS_ * D_ * 4);
  unsigned short* hb   = (unsigned short*)take((size_t)ROWS_ * D_ * 2);
  unsigned short* qb   = (unsigned short*)take((size_t)ROWS_ * D_ * 2);
  unsigned short* kb   = (unsigned short*)take((size_t)ROWS_ * D_ * 2);
  unsigned short* vt   = (unsigned short*)take((size_t)D_ * ROWS_ * 2);   // v transposed [256][32768]
  unsigned short* wqt  = (unsigned short*)take((size_t)D_ * D_ * 2);      // [N][K] transposed weights
  unsigned short* wkt  = (unsigned short*)take((size_t)D_ * D_ * 2);
  unsigned short* wvt  = (unsigned short*)take((size_t)D_ * D_ * 2);
  unsigned short* wot  = (unsigned short*)take((size_t)D_ * D_ * 2);
  unsigned short* w1t  = (unsigned short*)take((size_t)MLP_ * D_ * 2);    // [1024][256]
  unsigned short* w2t  = (unsigned short*)take((size_t)D_ * MLP_ * 2);    // [256][1024]
  unsigned short* pjt  = (unsigned short*)take((size_t)MP_ * DH_ * 2);    // [288][64] m-major
  float*          dash = (float*)         take((size_t)ROWS_ * LDM_ * 4); // reused q then k
  unsigned short* qp   = (unsigned short*)take((size_t)ROWS_ * LDM_ * 2);
  unsigned short* kpT  = (unsigned short*)take((size_t)B_ * H_ * MP_ * N_ * 2);
  float*          diagq= (float*)         take((size_t)ROWS_ * H_ * 4);
  float*          diagk= (float*)         take((size_t)ROWS_ * H_ * 4);
  float*          part = (float*)         take(1024 * 4);
  float*          gmax = (float*)         take(256);
  float*          ksum = (float*)         take((size_t)B_ * H_ * MP_ * 4);
  unsigned short* ctxt = (unsigned short*)take((size_t)B_ * H_ * DH_ * MP_ * 2); // [bh][64][288]
  float*          att  = (float*)         take((size_t)ROWS_ * D_ * 4);
  unsigned short* attb = (unsigned short*)take((size_t)ROWS_ * D_ * 2);
  float*          dinv = (float*)         take((size_t)ROWS_ * H_ * 4);
  unsigned short* ff1b = (unsigned short*)take((size_t)ROWS_ * MLP_ * 2);

  // x working copy
  hipMemcpyAsync(xbuf, x_in, (size_t)ROWS_ * D_ * sizeof(float),
                 hipMemcpyDeviceToDevice, stream);

  for (int l = 0; l < DEPTH_; ++l) {
    // ---- per-layer weight conversion to bf16, K-major (transposed) ----
    int nDD = D_ * D_, nDM = D_ * MLP_;
    k_cvtT<<<(nDD + 255) / 256, 256, 0, stream>>>(wq + (size_t)l * nDD, wqt, D_, D_);
    k_cvtT<<<(nDD + 255) / 256, 256, 0, stream>>>(wk + (size_t)l * nDD, wkt, D_, D_);
    k_cvtT<<<(nDD + 255) / 256, 256, 0, stream>>>(wv + (size_t)l * nDD, wvt, D_, D_);
    k_cvtT<<<(nDD + 255) / 256, 256, 0, stream>>>(wo + (size_t)l * nDD, wot, D_, D_);
    k_cvtT<<<(nDM + 255) / 256, 256, 0, stream>>>(w1 + (size_t)l * nDM, w1t, D_, MLP_);
    k_cvtT<<<(nDM + 255) / 256, 256, 0, stream>>>(w2 + (size_t)l * nDM, w2t, MLP_, D_);
    k_projT<<<(MP_ * DH_ + 255) / 256, 256, 0, stream>>>(proj + (size_t)l * MF_ * DH_, pjt);

    // ---- LN1 -> h (bf16) ----
    k_ln<<<ROWS_, 256, 0, stream>>>(xbuf, ln1w + l * D_, ln1b + l * D_, hb);

    // ---- QKV projections (normalizer folded into q,k; v written transposed) ----
    gemm(stream, hb, D_, wqt, D_, nullptr, nullptr, nullptr, qb, nullptr, 0, D_,
         ROWS_, D_, D_, NORMER, 0);
    gemm(stream, hb, D_, wkt, D_, nullptr, nullptr, nullptr, kb, nullptr, 0, D_,
         ROWS_, D_, D_, NORMER, 0);
    gemm(stream, hb, D_, wvt, D_, nullptr, nullptr, nullptr, nullptr, vt, ROWS_, D_,
         ROWS_, D_, D_, 1.0f, 0);

    // ---- diag terms ----
    k_diag<<<(ROWS_ * H_ + 255) / 256, 256, 0, stream>>>(qb, diagq);
    k_diag<<<(ROWS_ * H_ + 255) / 256, 256, 0, stream>>>(kb, diagk);

    // ---- query features: dash = qn @ proj' per head, then exp ----
    for (int h = 0; h < H_; ++h)
      gemm(stream, qb + h * DH_, D_, pjt, DH_, nullptr, nullptr,
           dash + h * MP_, nullptr, nullptr, 0, LDM_, ROWS_, MP_, DH_, 1.0f, 0);
    k_exp_q<<<ROWS_ * H_, 128, 0, stream>>>(dash, diagq, qp);

    // ---- key features: dash reused, global max, exp -> kpT ----
    for (int h = 0; h < H_; ++h)
      gemm(stream, kb + h * DH_, D_, pjt, DH_, nullptr, nullptr,
           dash + h * MP_, nullptr, nullptr, 0, LDM_, ROWS_, MP_, DH_, 1.0f, 0);
    k_maxp<<<1024, 256, 0, stream>>>(dash, part, (size_t)ROWS_ * LDM_);
    k_maxf<<<1, 256, 0, stream>>>(part, gmax, 1024);
    k_exp_k<<<ROWS_ * H_, 128, 0, stream>>>(dash, diagk, gmax, kpT);

    // ---- k_sum and context = kp' @ v per (b,h) (context written transposed) ----
    k_ksum<<<B_ * H_ * MP_, 256, 0, stream>>>(kpT, ksum);
    for (int bh = 0; bh < B_ * H_; ++bh) {
      int b = bh >> 2, h = bh & 3;
      gemm(stream,
           kpT + (size_t)bh * MP_ * N_, N_,
           vt + (size_t)(h * DH_) * ROWS_ + b * N_, ROWS_,
           nullptr, nullptr, nullptr, nullptr,
           ctxt + (size_t)bh * DH_ * MP_, MP_,
           0, MP_, DH_, N_, 1.0f, 0);
    }

    // ---- d_inv, out = qp @ context per (b,h) ----
    k_dinv<<<(ROWS_ * H_ + 255) / 256, 256, 0, stream>>>(qp, ksum, dinv);
    for (int bh = 0; bh < B_ * H_; ++bh) {
      int b = bh >> 2, h = bh & 3;
      gemm(stream,
           qp + (size_t)b * N_ * LDM_ + h * MP_, LDM_,
           ctxt + (size_t)bh * DH_ * MP_, MP_,
           nullptr, nullptr,
           att + (size_t)b * N_ * D_ + h * DH_, nullptr, nullptr, 0, D_,
           N_, DH_, MP_, 1.0f, 0);
    }
    k_scale<<<(ROWS_ * D_ + 255) / 256, 256, 0, stream>>>(att, dinv, attb);

    // ---- W_O + bo + residual -> x (in place) ----
    gemm(stream, attb, D_, wot, D_, bo + l * D_, xbuf, xbuf, nullptr, nullptr, 0, D_,
         ROWS_, D_, D_, 1.0f, 0);

    // ---- LN2 + MLP (gelu) + residual ----
    k_ln<<<ROWS_, 256, 0, stream>>>(xbuf, ln2w + l * D_, ln2b + l * D_, hb);
    gemm(stream, hb, D_, w1t, D_, b1 + (size_t)l * MLP_, nullptr,
         nullptr, ff1b, nullptr, 0, MLP_, ROWS_, MLP_, D_, 1.0f, 1);
    gemm(stream, ff1b, MLP_, w2t, MLP_, b2 + l * D_, xbuf, xbuf, nullptr, nullptr, 0, D_,
         ROWS_, D_, MLP_, 1.0f, 0);
  }

  hipMemcpyAsync(d_out, xbuf, (size_t)ROWS_ * D_ * sizeof(float),
                 hipMemcpyDeviceToDevice, stream);
}
